// mLSTMBlock_50706383897183
// MI455X (gfx1250) — compile-verified, matrix operand order checked by reading, not run on hardware
//
#include <hip/hip_runtime.h>

typedef __bf16 bf16_t;
typedef __bf16 v16bf __attribute__((ext_vector_type(16)));
typedef __bf16 v8bf  __attribute__((ext_vector_type(8)));
typedef float  v8f   __attribute__((ext_vector_type(8)));

#define LDSS 40           // LDS row stride in bf16 elems (80B rows: 16B-aligned, conflict-free)
#define TILE_BYTES 10240  // 128 * LDSS * sizeof(bf16)

__device__ __forceinline__ unsigned short f2bfbits(float f) {
  unsigned u = __float_as_uint(f);
  u += 0x7FFFu + ((u >> 16) & 1u);            // round-to-nearest-even
  return (unsigned short)(u >> 16);
}
__device__ __forceinline__ bf16_t f2bf(float f) {
  unsigned short h = f2bfbits(f);
  bf16_t r; __builtin_memcpy(&r, &h, 2); return r;
}
__device__ __forceinline__ float bf2f(bf16_t b) {
  unsigned short h; __builtin_memcpy(&h, &b, 2);
  return __uint_as_float(((unsigned)h) << 16);
}
__device__ __forceinline__ unsigned pack2bf(float a, float b) {
  return (unsigned)f2bfbits(a) | ((unsigned)f2bfbits(b) << 16);
}
__device__ __forceinline__ v16bf cat16(v8bf lo, v8bf hi) {
  return __builtin_shufflevector(lo, hi, 0,1,2,3,4,5,6,7,8,9,10,11,12,13,14,15);
}
// LDS byte offset of a __shared__ pointer: flat LDS address bits [31:0]
__device__ __forceinline__ unsigned lds_off(const void* p) {
  return (unsigned)(reinterpret_cast<uintptr_t>(p));
}
// CDNA5 async DMA: 16B global -> LDS, tracked by ASYNCcnt
__device__ __forceinline__ void async_b128(unsigned lds_byte_off, const void* gaddr) {
  asm volatile("global_load_async_to_lds_b128 %0, %1, off"
               : : "v"(lds_byte_off), "v"(gaddr) : "memory");
}
__device__ __forceinline__ void wait_async0() {
  asm volatile("s_wait_asynccnt 0x0" : : : "memory");
}

// ---------------------------------------------------------------------------
// Weight pre-pass: W f32 [K][N] -> Wt bf16 [N][K]  (once per launch; weights
// are reused by 64 M-tiles, so the GEMMs then read half the bytes and do no
// conversion VALU in the hot loop). LDS-staged for coalescing on both sides.
// ---------------------------------------------------------------------------
__global__ __launch_bounds__(256) void wconvert_kernel(
    const float* __restrict__ W, bf16_t* __restrict__ Wt, int K, int N)
{
  __shared__ float tile[64][65];
  const int z = blockIdx.z;
  W  += (size_t)z * K * N;
  Wt += (size_t)z * K * N;
  const int k0 = blockIdx.y * 64, n0 = blockIdx.x * 64;
  const int tn = threadIdx.x & 63, tk = threadIdx.x >> 6;
  #pragma unroll
  for (int p = 0; p < 16; ++p) {
    int k = tk + p * 4;
    tile[k][tn] = W[(size_t)(k0 + k) * N + n0 + tn];
  }
  __syncthreads();
  const int wk = threadIdx.x & 63, wn = threadIdx.x >> 6;
  #pragma unroll
  for (int p = 0; p < 16; ++p) {
    int n = wn + p * 4;
    Wt[(size_t)(n0 + n) * K + k0 + wk] = f2bf(tile[wk][n]);
  }
}

// ---------------------------------------------------------------------------
// bf16 WMMA GEMM, double-buffered async-to-LDS pipeline:
//   out = (A_bf16[M,K] @ Wt_bf16[N,K]^T + bias[N]) * alpha (+resid)
// Block: 256 thr = 8 waves; tile 128x128, K-step 32; wave tile 64x32 (4x2 WMMA)
// Tile t+1 streams in via GLOBAL_LOAD_ASYNC_TO_LDS_B128 while tile t does WMMA.
// grid.z batches the block-diagonal projections via element strides.
// ---------------------------------------------------------------------------
__global__ __launch_bounds__(256) void gemm_bf16_wmma(
    const bf16_t* __restrict__ A, const bf16_t* __restrict__ Wt,
    const float* __restrict__ bias, float* __restrict__ C,
    bf16_t* __restrict__ Cbf, const float* __restrict__ resid,
    float alpha, int K, int lda, int ldc,
    int aBatch, int wBatch, int cBatch, int bBatch)
{
  __shared__ __align__(16) bf16_t sA[2][128 * LDSS];
  __shared__ __align__(16) bf16_t sB[2][128 * LDSS];

  const int tid = threadIdx.x;
  const int z   = blockIdx.z;
  const int m0  = blockIdx.y * 128;
  const int n0  = blockIdx.x * 128;

  A    += (size_t)z * aBatch;
  Wt   += (size_t)z * wBatch;
  bias += (size_t)z * bBatch;
  if (C)     C     += (size_t)z * cBatch;
  if (Cbf)   Cbf   += (size_t)z * cBatch;
  if (resid) resid += (size_t)z * cBatch;

  const int lane  = tid & 31;
  const int wid   = tid >> 5;
  const int waveM = wid >> 2;      // 0..1
  const int waveN = wid & 3;       // 0..3
  const int lrow  = lane & 15;
  const int khalf = lane >> 4;

  // async copy assignment: thread -> (row 0..127, 16-elem half of the 32-k row)
  const int crow = tid >> 1;
  const int coff = (tid & 1) << 4;
  const bf16_t* gA = A  + (size_t)(m0 + crow) * lda + coff;
  const bf16_t* gB = Wt + (size_t)(n0 + crow) * K   + coff;
  const unsigned ldsA0 = lds_off(&sA[0][crow * LDSS + coff]);
  const unsigned ldsB0 = lds_off(&sB[0][crow * LDSS + coff]);

  auto issue_tile = [&](int kt, int buf) {
    const bf16_t* a = gA + kt;
    const bf16_t* b = gB + kt;
    const unsigned la = ldsA0 + (unsigned)buf * TILE_BYTES;
    const unsigned lb = ldsB0 + (unsigned)buf * TILE_BYTES;
    async_b128(la,       a);
    async_b128(la + 16u, a + 8);
    async_b128(lb,       b);
    async_b128(lb + 16u, b + 8);
  };

  v8f acc[4][2];
  #pragma unroll
  for (int i = 0; i < 4; ++i)
    #pragma unroll
    for (int j = 0; j < 2; ++j)
      #pragma unroll
      for (int r = 0; r < 8; ++r) acc[i][j][r] = 0.0f;

  const int nk = K >> 5;
  issue_tile(0, 0);

  for (int t = 0; t < nk; ++t) {
    const int cur = t & 1;
    wait_async0();        // this wave's copies of tile t have landed (in-order)
    __syncthreads();      // all waves' copies landed; nobody still reads buf cur^1
    if (t + 1 < nk) issue_tile((t + 1) << 5, cur ^ 1);  // overlap DMA with WMMA

    const bf16_t* bufA = sA[cur];
    const bf16_t* bufB = sB[cur];
    v16bf af[4], bfr[2];
    #pragma unroll
    for (int i = 0; i < 4; ++i) {
      const bf16_t* ap = &bufA[(waveM * 64 + i * 16 + lrow) * LDSS + khalf * 8];
      af[i] = cat16(*(const v8bf*)ap, *(const v8bf*)(ap + 16));
    }
    #pragma unroll
    for (int j = 0; j < 2; ++j) {
      const bf16_t* bp = &bufB[(waveN * 32 + j * 16 + lrow) * LDSS + khalf * 8];
      bfr[j] = cat16(*(const v8bf*)bp, *(const v8bf*)(bp + 16));
    }
    #pragma unroll
    for (int i = 0; i < 4; ++i)
      #pragma unroll
      for (int j = 0; j < 2; ++j)
        acc[i][j] = __builtin_amdgcn_wmma_f32_16x16x32_bf16(
            false, af[i], false, bfr[j], (short)0, acc[i][j], false, false);
  }

  // ---- epilogue: three straight-line variants, addresses stepped by ldc ----
  const int mBase = m0 + waveM * 64 + khalf * 8;
  const int nBase = n0 + waveN * 32 + lrow;
  if (resid != nullptr) {                       // f32 out + residual
    #pragma unroll
    for (int i = 0; i < 4; ++i) {
      #pragma unroll
      for (int j = 0; j < 2; ++j) {
        const int col = nBase + j * 16;
        const float bcol = bias[col];
        size_t idx = (size_t)(mBase + i * 16) * ldc + col;
        #pragma unroll
        for (int r = 0; r < 8; ++r, idx += ldc)
          C[idx] = (acc[i][j][r] + bcol) * alpha + resid[idx];
      }
    }
  } else if (C != nullptr) {                    // f32 out
    #pragma unroll
    for (int i = 0; i < 4; ++i) {
      #pragma unroll
      for (int j = 0; j < 2; ++j) {
        const int col = nBase + j * 16;
        const float bcol = bias[col];
        size_t idx = (size_t)(mBase + i * 16) * ldc + col;
        #pragma unroll
        for (int r = 0; r < 8; ++r, idx += ldc)
          C[idx] = (acc[i][j][r] + bcol) * alpha;
      }
    }
  } else {                                      // bf16 out
    #pragma unroll
    for (int i = 0; i < 4; ++i) {
      #pragma unroll
      for (int j = 0; j < 2; ++j) {
        const int col = nBase + j * 16;
        const float bcol = bias[col];
        size_t idx = (size_t)(mBase + i * 16) * ldc + col;
        #pragma unroll
        for (int r = 0; r < 8; ++r, idx += ldc)
          Cbf[idx] = f2bf((acc[i][j][r] + bcol) * alpha);
      }
    }
  }
}

// ---------------------------------------------------------------------------
// denom[h1] = max_h2 | sum_b n_t[b,h1] * q[b,h2] |   (H x B x H WMMA + max)
// A = n_t^T (transpose-convert on store), B = q (convert on store).
// atomicMax on nonneg float-as-uint.
// ---------------------------------------------------------------------------
__global__ __launch_bounds__(256) void denom_wmma_kernel(
    const float* __restrict__ nT, const float* __restrict__ q,
    float* __restrict__ denom, int K, int H)
{
  __shared__ __align__(16) bf16_t As[128 * LDSS];
  __shared__ __align__(16) bf16_t Bs[128 * LDSS];

  const int tid = threadIdx.x;
  const int m0 = blockIdx.y * 128, n0 = blockIdx.x * 128;
  const int lane = tid & 31, wid = tid >> 5;
  const int waveM = wid >> 2, waveN = wid & 3;
  const int lrow = lane & 15, khalf = lane >> 4;

  v8f acc[4][2];
  #pragma unroll
  for (int i = 0; i < 4; ++i)
    #pragma unroll
    for (int j = 0; j < 2; ++j)
      #pragma unroll
      for (int r = 0; r < 8; ++r) acc[i][j][r] = 0.0f;

  const int am4 = (tid & 31) * 4;   // m quad
  const int ak  = tid >> 5;         // k row 0..7
  const int bn  = tid & 127;
  const int bq  = tid >> 7;

  for (int kt = 0; kt < K; kt += 32) {
    #pragma unroll
    for (int p = 0; p < 4; ++p) {
      int k = ak + p * 8;
      float4 v = *(const float4*)(nT + (size_t)(kt + k) * H + m0 + am4);
      As[(am4 + 0) * LDSS + k] = f2bf(v.x);
      As[(am4 + 1) * LDSS + k] = f2bf(v.y);
      As[(am4 + 2) * LDSS + k] = f2bf(v.z);
      As[(am4 + 3) * LDSS + k] = f2bf(v.w);
    }
    #pragma unroll
    for (int p = 0; p < 4; ++p) {
      int k0 = (bq + p * 2) * 4;
      size_t base = (size_t)(kt + k0) * H + n0 + bn;
      uint2 pk;
      pk.x = pack2bf(q[base],                 q[base + (size_t)H]);
      pk.y = pack2bf(q[base + 2 * (size_t)H], q[base + 3 * (size_t)H]);
      *(uint2*)(&Bs[bn * LDSS + k0]) = pk;
    }
    __syncthreads();

    v16bf af[4], bfr[2];
    #pragma unroll
    for (int i = 0; i < 4; ++i) {
      const bf16_t* ap = &As[(waveM * 64 + i * 16 + lrow) * LDSS + khalf * 8];
      af[i] = cat16(*(const v8bf*)ap, *(const v8bf*)(ap + 16));
    }
    #pragma unroll
    for (int j = 0; j < 2; ++j) {
      const bf16_t* bp = &Bs[(waveN * 32 + j * 16 + lrow) * LDSS + khalf * 8];
      bfr[j] = cat16(*(const v8bf*)bp, *(const v8bf*)(bp + 16));
    }
    #pragma unroll
    for (int i = 0; i < 4; ++i)
      #pragma unroll
      for (int j = 0; j < 2; ++j)
        acc[i][j] = __builtin_amdgcn_wmma_f32_16x16x32_bf16(
            false, af[i], false, bfr[j], (short)0, acc[i][j], false, false);
    __syncthreads();
  }

  #pragma unroll
  for (int i = 0; i < 4; ++i) {
    #pragma unroll
    for (int r = 0; r < 8; ++r) {
      float vmax = fmaxf(fabsf(acc[i][0][r]), fabsf(acc[i][1][r]));
      #pragma unroll
      for (int off = 1; off < 16; off <<= 1)
        vmax = fmaxf(vmax, __shfl_xor(vmax, off));
      if ((lane & 15) == 0) {
        int row = m0 + waveM * 64 + i * 16 + khalf * 8 + r;
        atomicMax((unsigned int*)(denom + row), __float_as_uint(vmax));
      }
    }
  }
}

// ---------------------------------------------------------------------------
// LayerNorm (row of 1024 f32) -> bf16
// ---------------------------------------------------------------------------
__global__ __launch_bounds__(256) void layernorm_kernel(
    const float* __restrict__ x, const float* __restrict__ g,
    const float* __restrict__ b, bf16_t* __restrict__ out, int D)
{
  __shared__ float red[16];
  const int row = blockIdx.x, tid = threadIdx.x;
  const float* xr = x + (size_t)row * D;
  float4 v = *(const float4*)(xr + tid * 4);
  float s  = v.x + v.y + v.z + v.w;
  float s2 = v.x*v.x + v.y*v.y + v.z*v.z + v.w*v.w;
  #pragma unroll
  for (int off = 1; off < 32; off <<= 1) { s += __shfl_xor(s, off); s2 += __shfl_xor(s2, off); }
  if ((tid & 31) == 0) { red[tid >> 5] = s; red[8 + (tid >> 5)] = s2; }
  __syncthreads();
  if (tid == 0) {
    float ts = 0.f, ts2 = 0.f;
    for (int i = 0; i < 8; ++i) { ts += red[i]; ts2 += red[8 + i]; }
    red[0] = ts; red[8] = ts2;
  }
  __syncthreads();
  float mu  = red[0] / D;
  float var = red[8] / D - mu * mu;
  float rs  = rsqrtf(var + 1e-5f);
  int c = tid * 4;
  out[(size_t)row * D + c + 0] = f2bf((v.x - mu) * rs * g[c + 0] + b[c + 0]);
  out[(size_t)row * D + c + 1] = f2bf((v.y - mu) * rs * g[c + 1] + b[c + 1]);
  out[(size_t)row * D + c + 2] = f2bf((v.z - mu) * rs * g[c + 2] + b[c + 2]);
  out[(size_t)row * D + c + 3] = f2bf((v.w - mu) * rs * g[c + 3] + b[c + 3]);
}

// ---------------------------------------------------------------------------
// Causal conv1d (k=4, left pad 3) + SiLU, bf16 -> bf16
// ---------------------------------------------------------------------------
__global__ __launch_bounds__(256) void conv_silu_kernel(
    const bf16_t* __restrict__ xin, const float* __restrict__ w,
    const float* __restrict__ cb, bf16_t* __restrict__ out, int L)
{
  int l = blockIdx.x * 256 + threadIdx.x;
  int row = blockIdx.y;
  const bf16_t* xr = xin + (size_t)row * L;
  float acc = cb[0];
  #pragma unroll
  for (int j = 0; j < 4; ++j) {
    int ll = l - 3 + j;
    if (ll >= 0) acc += w[j] * bf2f(xr[ll]);
  }
  float sv = acc / (1.f + __expf(-acc));
  out[(size_t)row * L + l] = f2bf(sv);
}

// ---------------------------------------------------------------------------
// mLSTM gates (elementwise); sigmoid(o) written in place over o_pre
// ---------------------------------------------------------------------------
__global__ __launch_bounds__(256) void gates_kernel(
    const float* __restrict__ it_, const float* __restrict__ ft_,
    float* __restrict__ opre_io,
    const float* __restrict__ kv, const float* __restrict__ vv,
    const float* __restrict__ cprev, const float* __restrict__ nprev,
    const float* __restrict__ mprev,
    float* __restrict__ c_out, float* __restrict__ n_out,
    float* __restrict__ m_out, unsigned long long total)
{
  unsigned long long i = (unsigned long long)blockIdx.x * 256ull + threadIdx.x;
  if (i >= total) return;
  float it = it_[i], ft = ft_[i], mp = mprev[i];
  float mt = fmaxf(ft + mp, it);
  float ig = __expf(it - mt);
  float fg = __expf(ft + mp - mt);
  float k  = kv[i];
  c_out[i] = fg * cprev[i] + ig * (vv[i] * k);
  n_out[i] = fg * nprev[i] + ig * k;
  m_out[i] = mt;
  float op = opre_io[i];
  opre_io[i] = 1.f / (1.f + __expf(-op));
}

// ---------------------------------------------------------------------------
// h_t = o*c_t*q/denom[h]; groupnorm(8 groups of 256) + skip; * silu(upr)
// one wave exactly covers one 256-wide group (32 lanes x 8 elems)
// ---------------------------------------------------------------------------
__global__ __launch_bounds__(256) void hgn_kernel(
    const float* __restrict__ og, const float* __restrict__ ct,
    const float* __restrict__ qv, const float* __restrict__ denom,
    const float* __restrict__ gng, const float* __restrict__ gnb,
    const float* __restrict__ skip, const float* __restrict__ upr,
    float* __restrict__ h_out, bf16_t* __restrict__ outpre, int H)
{
  const int row = blockIdx.x, tid = threadIdx.x;
  const size_t base = (size_t)row * H;
  const int h0 = tid * 8;
  float hv[8];
  float s = 0.f, s2 = 0.f;
  #pragma unroll
  for (int j = 0; j < 8; ++j) {
    int h = h0 + j;
    float v = og[base + h] * ct[base + h] * qv[base + h] / denom[h];
    hv[j] = v; s += v; s2 += v * v;
    h_out[base + h] = v;
  }
  #pragma unroll
  for (int off = 1; off < 32; off <<= 1) { s += __shfl_xor(s, off); s2 += __shfl_xor(s2, off); }
  float mu  = s * (1.f / 256.f);
  float var = s2 * (1.f / 256.f) - mu * mu;
  float rs  = rsqrtf(var + 1e-5f);
  #pragma unroll
  for (int j = 0; j < 8; ++j) {
    int h = h0 + j;
    float o = (hv[j] - mu) * rs * gng[h] + gnb[h] + skip[base + h];
    float ur = upr[base + h];
    o *= ur / (1.f + __expf(-ur));
    outpre[base + h] = f2bf(o);
  }
}

__global__ __launch_bounds__(256) void zero_kernel(float* __restrict__ p, int n)
{
  int i = blockIdx.x * 256 + threadIdx.x;
  if (i < n) p[i] = 0.0f;
}

// ---------------------------------------------------------------------------
extern "C" void kernel_launch(void* const* d_in, const int* in_sizes, int n_in,
                              void* d_out, int out_size, void* d_ws, size_t ws_size,
                              hipStream_t stream)
{
  (void)in_sizes; (void)n_in; (void)out_size; (void)ws_size;
  const int B = 8192, D = 1024, H = 2048, UP = 2048, NB = 8;

  const float* x      = (const float*)d_in[0];
  const float* ln_g   = (const float*)d_in[1];
  const float* ln_b   = (const float*)d_in[2];
  const float* Wupl   = (const float*)d_in[3];
  const float* bupl   = (const float*)d_in[4];
  const float* Wupr   = (const float*)d_in[5];
  const float* bupr   = (const float*)d_in[6];
  const float* conv_w = (const float*)d_in[7];
  const float* conv_b = (const float*)d_in[8];
  const float* Wskip  = (const float*)d_in[9];
  const float* bskip  = (const float*)d_in[10];
  const float* Wq     = (const float*)d_in[11];
  const float* bq     = (const float*)d_in[12];
  const float* Wk     = (const float*)d_in[13];
  const float* bk     = (const float*)d_in[14];
  const float* Wv     = (const float*)d_in[15];
  const float* bv     = (const float*)d_in[16];
  const float* Wi     = (const float*)d_in[17];
  const float* bi     = (const float*)d_in[18];
  const float* Wf     = (const float*)d_in[19];
  const float* bfg    = (const float*)d_in[20];
  const float* Wo     = (const float*)d_in[21];
  const float* bo     = (const float*)d_in[22];
  const float* gn_g   = (const float*)d_in[23];
  const float* gn_b   = (const float*)d_in[24];
  const float* Wdown  = (const float*)d_in[25];
  const float* bdown  = (const float*)d_in[26];
  const float* c_prev = (const float*)d_in[27];
  const float* n_prev = (const float*)d_in[28];
  const float* m_prev = (const float*)d_in[29];

  // output tuple slices: (final, h_t, c_t, n_t, m_t) flat-concatenated
  float* out_final = (float*)d_out;
  float* out_h = out_final + (size_t)B * D;
  float* out_c = out_h + (size_t)B * H;
  float* out_n = out_c + (size_t)B * H;
  float* out_m = out_n + (size_t)B * H;

  // workspace carve-up (all 256B aligned)
  char* wp = (char*)d_ws;
  auto take = [&](size_t bytes) -> char* {
    char* p = wp; wp += (bytes + 255) & ~(size_t)255; return p;
  };
  bf16_t* xnorm  = (bf16_t*)take((size_t)B * D  * 2);
  bf16_t* upl    = (bf16_t*)take((size_t)B * UP * 2);
  bf16_t* xconv  = (bf16_t*)take((size_t)B * UP * 2);
  bf16_t* outpre = (bf16_t*)take((size_t)B * H  * 2);
  float*  upr    = (float*)take((size_t)B * H * 4);
  float*  skip   = (float*)take((size_t)B * H * 4);
  float*  qv     = (float*)take((size_t)B * H * 4);
  float*  kv     = (float*)take((size_t)B * H * 4);
  float*  vv     = (float*)take((size_t)B * H * 4);
  float*  itl    = (float*)take((size_t)B * H * 4);
  float*  ftl    = (float*)take((size_t)B * H * 4);
  float*  opre   = (float*)take((size_t)B * H * 4);
  float*  denom  = (float*)take((size_t)H * 4);
  // pre-transposed bf16 weights [N][K]
  bf16_t* WuplT  = (bf16_t*)take((size_t)D * UP * 2);
  bf16_t* WuprT  = (bf16_t*)take((size_t)D * H  * 2);
  bf16_t* WskipT = (bf16_t*)take((size_t)UP * H * 2);
  bf16_t* WiT    = (bf16_t*)take((size_t)UP * H * 2);
  bf16_t* WfT    = (bf16_t*)take((size_t)UP * H * 2);
  bf16_t* WoT    = (bf16_t*)take((size_t)UP * H * 2);
  bf16_t* WdownT = (bf16_t*)take((size_t)H * D  * 2);
  bf16_t* WqT    = (bf16_t*)take((size_t)NB * 256 * 256 * 2);
  bf16_t* WkT    = (bf16_t*)take((size_t)NB * 256 * 256 * 2);
  bf16_t* WvT    = (bf16_t*)take((size_t)NB * 256 * 256 * 2);

  dim3 blk(256);

  // 0. weight convert+transpose pre-pass (f32 [K][N] -> bf16 [N][K])
  wconvert_kernel<<<dim3(UP/64, D/64,  1), blk, 0, stream>>>(Wupl,  WuplT,  D,  UP);
  wconvert_kernel<<<dim3(H/64,  D/64,  1), blk, 0, stream>>>(Wupr,  WuprT,  D,  H);
  wconvert_kernel<<<dim3(H/64,  UP/64, 1), blk, 0, stream>>>(Wskip, WskipT, UP, H);
  wconvert_kernel<<<dim3(H/64,  UP/64, 1), blk, 0, stream>>>(Wi,    WiT,    UP, H);
  wconvert_kernel<<<dim3(H/64,  UP/64, 1), blk, 0, stream>>>(Wf,    WfT,    UP, H);
  wconvert_kernel<<<dim3(H/64,  UP/64, 1), blk, 0, stream>>>(Wo,    WoT,    UP, H);
  wconvert_kernel<<<dim3(D/64,  H/64,  1), blk, 0, stream>>>(Wdown, WdownT, H,  D);
  wconvert_kernel<<<dim3(4, 4, NB), blk, 0, stream>>>(Wq, WqT, 256, 256);
  wconvert_kernel<<<dim3(4, 4, NB), blk, 0, stream>>>(Wk, WkT, 256, 256);
  wconvert_kernel<<<dim3(4, 4, NB), blk, 0, stream>>>(Wv, WvT, 256, 256);

  // 1. layernorm -> bf16
  layernorm_kernel<<<dim3(B), blk, 0, stream>>>(x, ln_g, ln_b, xnorm, D);
  // 2. x_up_left = xnorm @ Wupl + bupl (bf16 out)
  gemm_bf16_wmma<<<dim3(UP/128, B/128, 1), blk, 0, stream>>>(
      xnorm, WuplT, bupl, nullptr, upl, nullptr, 1.f, D, D, UP, 0, 0, 0, 0);
  // 3. x_up_right = xnorm @ Wupr + bupr (f32)
  gemm_bf16_wmma<<<dim3(H/128, B/128, 1), blk, 0, stream>>>(
      xnorm, WuprT, bupr, upr, nullptr, nullptr, 1.f, D, D, H, 0, 0, 0, 0);
  // 4. causal conv + silu
  conv_silu_kernel<<<dim3(UP/256, B), blk, 0, stream>>>(upl, conv_w, conv_b, xconv, UP);
  // 5. skip = xconv @ Wskip + bskip
  gemm_bf16_wmma<<<dim3(H/128, B/128, 1), blk, 0, stream>>>(
      xconv, WskipT, bskip, skip, nullptr, nullptr, 1.f, UP, UP, H, 0, 0, 0, 0);
  // 6-8. block-diagonal q, k (scaled 1/sqrt(256)), v  (grid.z = 8 blocks)
  gemm_bf16_wmma<<<dim3(2, B/128, NB), blk, 0, stream>>>(
      xconv, WqT, bq, qv, nullptr, nullptr, 1.f,      256, UP, H, 256, 65536, 256, 256);
  gemm_bf16_wmma<<<dim3(2, B/128, NB), blk, 0, stream>>>(
      xconv, WkT, bk, kv, nullptr, nullptr, 1.f/16.f, 256, UP, H, 256, 65536, 256, 256);
  gemm_bf16_wmma<<<dim3(2, B/128, NB), blk, 0, stream>>>(
      upl,   WvT, bv, vv, nullptr, nullptr, 1.f,      256, UP, H, 256, 65536, 256, 256);
  // 9-11. i_tilde, f_tilde, o_pre
  gemm_bf16_wmma<<<dim3(H/128, B/128, 1), blk, 0, stream>>>(
      xconv, WiT, bi,  itl,  nullptr, nullptr, 1.f, UP, UP, H, 0, 0, 0, 0);
  gemm_bf16_wmma<<<dim3(H/128, B/128, 1), blk, 0, stream>>>(
      xconv, WfT, bfg, ftl,  nullptr, nullptr, 1.f, UP, UP, H, 0, 0, 0, 0);
  gemm_bf16_wmma<<<dim3(H/128, B/128, 1), blk, 0, stream>>>(
      upl,   WoT, bo,  opre, nullptr, nullptr, 1.f, UP, UP, H, 0, 0, 0, 0);
  // 12. gates -> c_t, n_t, m_t (into d_out), sigmoid(o) in place
  unsigned long long total = (unsigned long long)B * H;
  gates_kernel<<<dim3((unsigned)((total + 255) / 256)), blk, 0, stream>>>(
      itl, ftl, opre, kv, vv, c_prev, n_prev, m_prev, out_c, out_n, out_m, total);
  // 13-14. denom = max_h2 |n_t^T @ q|
  zero_kernel<<<dim3((H + 255) / 256), blk, 0, stream>>>(denom, H);
  denom_wmma_kernel<<<dim3(H/128, H/128), blk, 0, stream>>>(out_n, qv, denom, B, H);
  // 15. h_t, groupnorm + skip, * silu(upr) -> bf16
  hgn_kernel<<<dim3(B), blk, 0, stream>>>(
      opre, out_c, qv, denom, gn_g, gn_b, skip, upr, out_h, outpre, H);
  // 16. final = outpre @ Wdown + bdown + x
  gemm_bf16_wmma<<<dim3(D/128, B/128, 1), blk, 0, stream>>>(
      outpre, WdownT, bdown, out_final, nullptr, x, 1.f, H, H, D, 0, 0, 0, 0);
}
